// ResTransformer_71734543778233
// MI455X (gfx1250) — matrix operand order repaired and checked by reading, not compiled, not run on hardware
//
#include <hip/hip_runtime.h>
#include <math.h>

typedef __attribute__((ext_vector_type(16))) _Float16 v16h;
typedef __attribute__((ext_vector_type(8)))  _Float16 v8h;
typedef __attribute__((ext_vector_type(8)))  float    v8f;

#define HID 64

// ---------- monotonic float<->uint key for atomicMax on signed floats ----------
__device__ __forceinline__ unsigned fkey(float f) {
    unsigned u = __float_as_uint(f);
    return (u & 0x80000000u) ? ~u : (u | 0x80000000u);
}
__device__ __forceinline__ float funkey(unsigned k) {
    return __uint_as_float((k & 0x80000000u) ? (k & 0x7FFFFFFFu) : ~k);
}

// ---------- elementwise log1p ----------
__global__ void k_log1p(const float* __restrict__ in, float* __restrict__ out, int n) {
    int i = blockIdx.x * blockDim.x + threadIdx.x;
    if (i < n) out[i] = logf(in[i] + 1.0f);
}

// ---------- reset per-node softmax state ----------
__global__ void k_reset(unsigned* __restrict__ mkey, float* __restrict__ ssum, int n) {
    int i = blockIdx.x * blockDim.x + threadIdx.x;
    if (i < n) { mkey[i] = 0u; ssum[i] = 0.0f; }
}

// ---------- pack f32 weights [7,64,64] into f16 WMMA B-fragment lane layout ----------
// Fragment f = m*8 + ct*2 + ko  (m: matrix 0..3, ct: col tile 0..3, ko: K chunk 0..1).
// Lane holds 16 halves: col = ct*16 + lane%16, K = ko*32 + 16*(lane/16) + i.
__global__ void k_pack_w(const float* __restrict__ Wq, const float* __restrict__ Wk,
                         const float* __restrict__ Wv, const float* __restrict__ Ws,
                         _Float16* __restrict__ out, int layers) {
    int gid = blockIdx.x * blockDim.x + threadIdx.x;
    int lane = gid & 31;
    int f    = (gid >> 5) & 31;
    int l    = gid >> 10;
    if (l >= layers) return;
    int m = f >> 3, ct = (f >> 1) & 3, ko = f & 1;
    const float* W = (m == 0) ? Wq : (m == 1) ? Wk : (m == 2) ? Wv : Ws;
    W += (size_t)l * 64 * 64;
    int ncol  = ct * 16 + (lane & 15);
    int kBase = ko * 32 + 16 * (lane >> 4);
    _Float16* dst = out + ((size_t)(l * 32 + f) * 32 + lane) * 16;
#pragma unroll
    for (int i = 0; i < 16; i++) dst[i] = (_Float16)W[ncol * 64 + kBase + i];
}

// ---------- layer 1 node transform: 11 -> 64, scalar (K too small for WMMA) ----------
__global__ void k_l1_node(const float* __restrict__ h0,                 // [N,11]
                          const float* __restrict__ Wq, const float* __restrict__ Wk,
                          const float* __restrict__ Wv, const float* __restrict__ Ws, // [64,11]
                          const float* __restrict__ bq, const float* __restrict__ bk,
                          const float* __restrict__ bv, const float* __restrict__ bs,
                          float* __restrict__ q, float* __restrict__ k,
                          float* __restrict__ v, float* __restrict__ y, int N) {
    int gid = blockIdx.x * blockDim.x + threadIdx.x;
    int n = gid >> 6, c = gid & 63;
    if (n >= N) return;
    float hr[11];
#pragma unroll
    for (int d = 0; d < 11; d++) hr[d] = h0[n * 11 + d];
    float aq = bq[c], ak = bk[c], av = bv[c], as_ = bs[c];
#pragma unroll
    for (int d = 0; d < 11; d++) {
        aq  = fmaf(hr[d], Wq[c * 11 + d], aq);
        ak  = fmaf(hr[d], Wk[c * 11 + d], ak);
        av  = fmaf(hr[d], Wv[c * 11 + d], av);
        as_ = fmaf(hr[d], Ws[c * 11 + d], as_);
    }
    q[n * 64 + c] = aq; k[n * 64 + c] = ak; v[n * 64 + c] = av; y[n * 64 + c] = as_;
}

// ---------- WMMA node GEMM over FULL 16-row tiles only (no store guards) ----------
// hin: f16 [N,64]; wl: pre-packed f16 fragments for this layer (32 frags x 32 lanes x 16 halves)
__global__ __launch_bounds__(256) void k_gemm_qkvs(
    const _Float16* __restrict__ hin, const _Float16* __restrict__ wl,
    const float* __restrict__ bq, const float* __restrict__ bk,
    const float* __restrict__ bv, const float* __restrict__ bs,
    float* __restrict__ q, float* __restrict__ k,
    float* __restrict__ v, float* __restrict__ y, int nTiles) {
    int wave = blockIdx.x * (blockDim.x >> 5) + (threadIdx.x >> 5);
    if (wave >= nTiles) return;               // wave-uniform: EXEC stays all-ones
    int lane = threadIdx.x & 31;
    int half = lane >> 4;
    int l16  = lane & 15;
    int rowBase = wave * 16;

    // A fragments (16x32 f16): lanes 0-15 row M=lane, K 0-15; lanes 16-31 K 16-31
    const _Float16* pa = hin + (size_t)(rowBase + l16) * 64 + 16 * half;
    v8h a0lo = *(const v8h*)(pa);
    v8h a0hi = *(const v8h*)(pa + 8);
    v8h a1lo = *(const v8h*)(pa + 32);
    v8h a1hi = *(const v8h*)(pa + 40);
    v16h a0 = __builtin_shufflevector(a0lo, a0hi, 0,1,2,3,4,5,6,7,8,9,10,11,12,13,14,15);
    v16h a1 = __builtin_shufflevector(a1lo, a1hi, 0,1,2,3,4,5,6,7,8,9,10,11,12,13,14,15);

    const v16h* wp = (const v16h*)wl;         // 32 x v16h per fragment
    const float* Bm[4] = {bq, bk, bv, bs};
    float*       Om[4] = {q, k, v, y};

#pragma unroll
    for (int m = 0; m < 4; m++) {
#pragma unroll
        for (int ct = 0; ct < 4; ct++) {
            int ncol = ct * 16 + l16;
            float bias = Bm[m][ncol];
            v8f c;
#pragma unroll
            for (int r = 0; r < 8; r++) c[r] = bias;  // bias per column; col fixed per lane
            v16h b0 = wp[(size_t)(m * 8 + ct * 2 + 0) * 32 + lane];
            v16h b1 = wp[(size_t)(m * 8 + ct * 2 + 1) * 32 + lane];
            c = __builtin_amdgcn_wmma_f32_16x16x32_f16(false, a0, false, b0, (short)0, c, false, false);
            c = __builtin_amdgcn_wmma_f32_16x16x32_f16(false, a1, false, b1, (short)0, c, false, false);
            float* o = Om[m] + (size_t)(rowBase + 8 * half) * 64 + ncol;
#pragma unroll
            for (int r = 0; r < 8; r++) o[(size_t)r * 64] = c[r];
        }
    }
}

// ---------- scalar tail for the (N % 16) leftover rows (usually zero) ----------
__global__ void k_gemm_tail(const _Float16* __restrict__ hin,
                            const float* __restrict__ Wq, const float* __restrict__ Wk,
                            const float* __restrict__ Wv, const float* __restrict__ Ws,
                            const float* __restrict__ bq, const float* __restrict__ bk,
                            const float* __restrict__ bv, const float* __restrict__ bs,
                            float* __restrict__ q, float* __restrict__ k,
                            float* __restrict__ v, float* __restrict__ y,
                            int rowStart, int N) {
    int gid = blockIdx.x * blockDim.x + threadIdx.x;
    int n = rowStart + (gid >> 6), c = gid & 63;
    if (n >= N) return;
    float aq = bq[c], ak = bk[c], av = bv[c], as_ = bs[c];
    for (int d = 0; d < 64; d++) {
        float x = (float)hin[(size_t)n * 64 + d];
        aq  = fmaf(x, Wq[c * 64 + d], aq);
        ak  = fmaf(x, Wk[c * 64 + d], ak);
        av  = fmaf(x, Wv[c * 64 + d], av);
        as_ = fmaf(x, Ws[c * 64 + d], as_);
    }
    q[(size_t)n * 64 + c] = aq; k[(size_t)n * 64 + c] = ak;
    v[(size_t)n * 64 + c] = av; y[(size_t)n * 64 + c] = as_;
}

// ---------- edge pass A: alpha = dot(q[dst], k[src]+e)/8 ; segment max ----------
__global__ void k_edge_alpha(const int* __restrict__ ei, const float* __restrict__ ea,
                             const float* __restrict__ We,              // [64,2]
                             const float* __restrict__ q, const float* __restrict__ k,
                             float* __restrict__ alpha, unsigned* __restrict__ mkey, int E) {
    int e = blockIdx.x * blockDim.x + threadIdx.x;
    if (e >= E) return;
    int src = ei[e], dst = ei[E + e];
    float a0 = ea[2 * e], a1 = ea[2 * e + 1];
    const float* qp = q + (size_t)dst * 64;
    const float* kp = k + (size_t)src * 64;
    float acc = 0.0f;
#pragma unroll
    for (int c = 0; c < 64; c++) {
        float ev = fmaf(a1, We[2 * c + 1], a0 * We[2 * c]);
        acc = fmaf(qp[c], kp[c] + ev, acc);
    }
    acc *= 0.125f;  // 1/sqrt(64)
    alpha[e] = acc;
    atomicMax(mkey + dst, fkey(acc));
}

// ---------- edge pass B: a = exp(alpha - m[dst]) ; segment sum ----------
__global__ void k_edge_exp(const int* __restrict__ ei, float* __restrict__ alpha,
                           const unsigned* __restrict__ mkey, float* __restrict__ ssum, int E) {
    int e = blockIdx.x * blockDim.x + threadIdx.x;
    if (e >= E) return;
    int dst = ei[E + e];
    unsigned km = mkey[dst];
    float m = (km == 0u) ? 0.0f : funkey(km);
    float a = expf(alpha[e] - m);
    alpha[e] = a;
    atomicAdd(ssum + dst, a);
}

// ---------- edge pass C: y[dst] += (v[src]+e) * a/(s[dst]+eps) ; 4 threads/edge ----------
__global__ void k_edge_scatter(const int* __restrict__ ei, const float* __restrict__ ea,
                               const float* __restrict__ We,
                               const float* __restrict__ v, const float* __restrict__ alpha,
                               const float* __restrict__ ssum, float* __restrict__ y, int E) {
    int gid = blockIdx.x * blockDim.x + threadIdx.x;
    int e = gid >> 2;
    if (e >= E) return;
    int cg = (gid & 3) * 16;
    int src = ei[e], dst = ei[E + e];
    float w = alpha[e] / (ssum[dst] + 1e-16f);
    float a0 = ea[2 * e], a1 = ea[2 * e + 1];
    const float* vp = v + (size_t)src * 64 + cg;
    float* yp = y + (size_t)dst * 64 + cg;
#pragma unroll
    for (int c = 0; c < 16; c++) {
        float ev = fmaf(a1, We[2 * (cg + c) + 1], a0 * We[2 * (cg + c)]);
        atomicAdd(yp + c, (vp[c] + ev) * w);
    }
}

// ---------- per-node std (ddof=1) + relu(h/std), emits f16 for the WMMA GEMM ----------
__global__ void k_std_relu(const float* __restrict__ h, _Float16* __restrict__ h2, int N) {
    int n = blockIdx.x * blockDim.x + threadIdx.x;
    if (n >= N) return;
    const float* p = h + (size_t)n * 64;
    float s = 0.0f, ss = 0.0f;
#pragma unroll
    for (int c = 0; c < 64; c++) { float x = p[c]; s += x; ss = fmaf(x, x, ss); }
    float var = (ss - s * s * (1.0f / 64.0f)) * (1.0f / 63.0f);
    float inv = 1.0f / sqrtf(var);
#pragma unroll
    for (int c0 = 0; c0 < 64; c0 += 8) {
        v8h t;
#pragma unroll
        for (int j = 0; j < 8; j++) {
            float x = p[c0 + j] * inv;
            t[j] = (_Float16)(x > 0.0f ? x : 0.0f);
        }
        *(v8h*)(h2 + (size_t)n * 64 + c0) = t;
    }
}

// ---------- residual add ----------
__global__ void k_resadd(float* __restrict__ h, const float* __restrict__ y, int n) {
    int i = blockIdx.x * blockDim.x + threadIdx.x;
    if (i < n) h[i] += y[i];
}

// ---------- final: out = (h/sqrt(7)) @ Wl^T + bl ----------
__global__ void k_final(const float* __restrict__ h, const float* __restrict__ Wl,
                        const float* __restrict__ bl, float* __restrict__ out, int N) {
    int n = blockIdx.x * blockDim.x + threadIdx.x;
    if (n >= N) return;
    const float scale = 0.3779644730092272f;  // 1/sqrt(7)
    const float* p = h + (size_t)n * 64;
    float acc = 0.0f;
#pragma unroll
    for (int c = 0; c < 64; c++) acc = fmaf(p[c], Wl[c], acc);
    out[n] = fmaf(acc, scale, bl[0]);
}

extern "C" void kernel_launch(void* const* d_in, const int* in_sizes, int n_in,
                              void* d_out, int out_size, void* d_ws, size_t ws_size,
                              hipStream_t stream) {
    const float* x     = (const float*)d_in[0];
    const int*   ei    = (const int*)d_in[1];
    const float* eattr = (const float*)d_in[2];
    const float* Wq1 = (const float*)d_in[3];
    const float* Wk1 = (const float*)d_in[4];
    const float* Wv1 = (const float*)d_in[5];
    const float* We1 = (const float*)d_in[6];
    const float* Ws1 = (const float*)d_in[7];
    const float* bq1 = (const float*)d_in[8];
    const float* bk1 = (const float*)d_in[9];
    const float* bv1 = (const float*)d_in[10];
    const float* bs1 = (const float*)d_in[11];
    const float* Wq  = (const float*)d_in[12];  // [7,64,64]
    const float* Wk  = (const float*)d_in[13];
    const float* Wv  = (const float*)d_in[14];
    const float* We  = (const float*)d_in[15];  // [7,64,2]
    const float* Ws  = (const float*)d_in[16];
    const float* bq  = (const float*)d_in[17];  // [7,64]
    const float* bk  = (const float*)d_in[18];
    const float* bv  = (const float*)d_in[19];
    const float* bs  = (const float*)d_in[20];
    const float* Wl  = (const float*)d_in[21];
    const float* bl  = (const float*)d_in[22];
    float* out = (float*)d_out;

    const int N = in_sizes[0] / 11;   // 50000
    const int E = in_sizes[2] / 2;    // 800000
    const int nTilesFull = N / 16;    // 3125 (exact here)
    const int remRows    = N - nTilesFull * 16;

    // carve workspace (256B-aligned blocks)
    char* wsp = (char*)d_ws;
    auto carve = [&](size_t bytes) -> void* {
        void* p = (void*)wsp;
        wsp += (bytes + 255) & ~(size_t)255;
        return p;
    };
    float*     h     = (float*)carve((size_t)N * 64 * 4);
    _Float16*  h2    = (_Float16*)carve((size_t)N * 64 * 2);
    float*     qb    = (float*)carve((size_t)N * 64 * 4);
    float*     kb    = (float*)carve((size_t)N * 64 * 4);
    float*     vb    = (float*)carve((size_t)N * 64 * 4);
    float*     yb    = (float*)carve((size_t)N * 64 * 4);
    float*     alpha = (float*)carve((size_t)E * 4);
    unsigned*  mkey  = (unsigned*)carve((size_t)N * 4);
    float*     ssum  = (float*)carve((size_t)N * 4);
    float*     h0    = (float*)carve((size_t)N * 11 * 4);
    float*     ea    = (float*)carve((size_t)E * 2 * 4);
    _Float16*  wpack = (_Float16*)carve((size_t)7 * 32 * 512 * 2);  // 7 layers x 32 frags x 512 halves

    const int T = 256;
    auto blk = [&](long n) { return (int)((n + T - 1) / T); };

    // prep: log1p + weight pre-pack (f16 fragment layout)
    k_log1p<<<blk((long)N * 11), T, 0, stream>>>(x, h0, N * 11);
    k_log1p<<<blk((long)E * 2),  T, 0, stream>>>(eattr, ea, E * 2);
    k_pack_w<<<blk(7L * 32 * 32), T, 0, stream>>>(Wq, Wk, Wv, Ws, wpack, 7);

    // ---- layer 1 (input conv, K=11) -> writes conv output into h ----
    k_reset<<<blk(N), T, 0, stream>>>(mkey, ssum, N);
    k_l1_node<<<blk((long)N * 64), T, 0, stream>>>(h0, Wq1, Wk1, Wv1, Ws1,
                                                   bq1, bk1, bv1, bs1,
                                                   qb, kb, vb, h, N);
    k_edge_alpha<<<blk(E), T, 0, stream>>>(ei, ea, We1, qb, kb, alpha, mkey, E);
    k_edge_exp<<<blk(E), T, 0, stream>>>(ei, alpha, mkey, ssum, E);
    k_edge_scatter<<<blk((long)E * 4), T, 0, stream>>>(ei, ea, We1, vb, alpha, ssum, h, E);

    // ---- 7 residual blocks (WMMA node GEMMs) ----
    const int gemmBlocks = (nTilesFull + 7) / 8;  // 8 waves per 256-thread block
    for (int b = 0; b < 7; b++) {
        const _Float16* wl = wpack + (size_t)b * 32 * 512;
        const float* pWe = We + (size_t)b * 64 * 2;
        const float* pbq = bq + (size_t)b * 64;
        const float* pbk = bk + (size_t)b * 64;
        const float* pbv = bv + (size_t)b * 64;
        const float* pbs = bs + (size_t)b * 64;

        k_std_relu<<<blk(N), T, 0, stream>>>(h, h2, N);
        k_reset<<<blk(N), T, 0, stream>>>(mkey, ssum, N);
        k_gemm_qkvs<<<gemmBlocks, T, 0, stream>>>(h2, wl, pbq, pbk, pbv, pbs,
                                                  qb, kb, vb, yb, nTilesFull);
        if (remRows > 0) {
            k_gemm_tail<<<blk((long)remRows * 64), T, 0, stream>>>(
                h2, Wq + (size_t)b * 64 * 64, Wk + (size_t)b * 64 * 64,
                Wv + (size_t)b * 64 * 64, Ws + (size_t)b * 64 * 64,
                pbq, pbk, pbv, pbs, qb, kb, vb, yb, nTilesFull * 16, N);
        }
        k_edge_alpha<<<blk(E), T, 0, stream>>>(ei, ea, pWe, qb, kb, alpha, mkey, E);
        k_edge_exp<<<blk(E), T, 0, stream>>>(ei, alpha, mkey, ssum, E);
        k_edge_scatter<<<blk((long)E * 4), T, 0, stream>>>(ei, ea, pWe, vb, alpha, ssum, yb, E);
        k_resadd<<<blk((long)N * 64), T, 0, stream>>>(h, yb, N * 64);
    }

    // ---- final projection ----
    k_final<<<blk(N), T, 0, stream>>>(h, Wl, bl, out, N);
}
